// Attention_867583393955
// MI455X (gfx1250) — matrix operand order repaired and checked by reading, not compiled, not run on hardware
//
#include <hip/hip_runtime.h>
#include <hip/hip_bf16.h>

// ---------------------------------------------------------------------------
// Problem constants (from reference)
// ---------------------------------------------------------------------------
static constexpr int Bx = 4;
static constexpr int Sx = 2048;
static constexpr int Ex = 1024;
static constexpr int Hx = 16;
static constexpr int Dx = 64;
static constexpr int Mx = Bx * Sx;          // 8192 rows for projection GEMMs
static constexpr float SCALE = 0.125f;      // 1/sqrt(64), folded into Q

typedef __attribute__((ext_vector_type(16))) __bf16 v16bf;
typedef __attribute__((ext_vector_type(8)))  __bf16 v8bf;
typedef __attribute__((ext_vector_type(8)))  float  v8f;
typedef __attribute__((ext_vector_type(4)))  int    v4i;

union V16Pack { v16bf v; v8bf h[2]; };

// ---------------------------------------------------------------------------
// Async global->LDS staging (GLOBAL_LOAD_ASYNC_TO_LDS_B128, ASYNCcnt).
// ASYNC ops complete in order -> double buffering waits on asynccnt<=N where
// N = ops just issued for the *next* tile, guaranteeing the current tile is
// resident. Guarded with a register-staging fallback.
// ---------------------------------------------------------------------------
#if defined(__has_builtin)
#  if __has_builtin(__builtin_amdgcn_global_load_async_to_lds_b128)
#    define HAVE_ASYNC_LDS 1
#  endif
#endif
#ifndef HAVE_ASYNC_LDS
#  define HAVE_ASYNC_LDS 0
#endif

__device__ __forceinline__ void stage16(const __bf16* g, __bf16* l) {
#if HAVE_ASYNC_LDS
  __builtin_amdgcn_global_load_async_to_lds_b128(
      (__attribute__((address_space(1))) v4i*)g,
      (__attribute__((address_space(3))) v4i*)l, 0, 0);
#else
  *(v8bf*)l = *(const v8bf*)g;
#endif
}

template <int N>
__device__ __forceinline__ void async_wait() {
#if HAVE_ASYNC_LDS
#  if defined(__has_builtin) && __has_builtin(__builtin_amdgcn_s_wait_asynccnt)
  __builtin_amdgcn_s_wait_asynccnt(N);
#  else
  asm volatile("s_wait_asynccnt %0" :: "i"(N) : "memory");
#  endif
#endif
}

__device__ __forceinline__ v8f wmma_bf16(v16bf a, v16bf b, v8f c) {
  return __builtin_amdgcn_wmma_f32_16x16x32_bf16(false, a, false, b,
                                                 (short)0, c, false, false);
}

// ---------------------------------------------------------------------------
// fp32 -> bf16 conversion (weights + activations); bandwidth-trivial
// ---------------------------------------------------------------------------
__global__ void f32_to_bf16_kernel(const float* __restrict__ src,
                                   __bf16* __restrict__ dst, int n) {
  int i = blockIdx.x * blockDim.x + threadIdx.x;
  if (i < n) dst[i] = (__bf16)src[i];
}

// ---------------------------------------------------------------------------
// WMMA GEMM: C[m,n] = sum_k A[m,k] * W[n,k] + bias[n]
//   A: [Mdim,Kdim] bf16 row-major, W: [Ndim,Kdim] bf16 row-major.
// Block = 8 waves sharing a 64-row M block; the 64x32 A tile is staged to
// LDS via async-to-LDS with double buffering (loads of tile i+1 overlap
// WMMAs of tile i). Each wave owns a 64x64 output tile: 4 A-matrices x
// 4 B-tiles = 16 WMMAs per K-step of 32.
// ---------------------------------------------------------------------------
template <bool OUT_BF16>
__global__ __launch_bounds__(256)
void gemm_bf16_nt_kernel(const __bf16* __restrict__ A,
                         const __bf16* __restrict__ W,
                         const float* __restrict__ bias,
                         void* __restrict__ Cout,
                         int Mdim, int Ndim, int Kdim) {
  __shared__ __bf16 As[2][64][32];              // 8KB ping-pong A tiles

  const int tid  = threadIdx.x;
  const int lane = tid & 31;
  const int wid  = tid >> 5;
  const int ntiles = Ndim >> 6;                 // 16: 8 waves/block share mt
  const int wave = blockIdx.x * 8 + wid;
  const int m0 = (wave / ntiles) << 6;
  const int n0 = (wave % ntiles) << 6;
  const int half = lane >> 4;
  const int l16  = lane & 15;
  const int aoff = half * 8;

  // staging map: 64 rows x 32 cols = 256 x 16B chunks, one per thread
  const int srow = tid >> 2;
  const int sch  = (tid & 3) * 8;
  const __bf16* agp = A + (size_t)(m0 + srow) * Kdim + sch;

  v8f acc[4][4] = {};
  const int NIT = Kdim >> 5;

  stage16(agp, &As[0][srow][sch]);              // prologue: tile 0

  for (int it = 0; it < NIT; ++it) {
    const int k0  = it << 5;
    const int cur = it & 1;
    if (it + 1 < NIT) {                         // overlap next tile's loads
      stage16(agp + k0 + 32, &As[cur ^ 1][srow][sch]);
      async_wait<1>();                          // current tile resident
    } else {
      async_wait<0>();
    }
    __syncthreads();

    v16bf a[4];
#pragma unroll
    for (int i = 0; i < 4; ++i) {
      const __bf16* ap = &As[cur][i * 16 + l16][aoff];
      V16Pack u;
      u.h[0] = *(const v8bf*)ap;                // K {off..off+7}
      u.h[1] = *(const v8bf*)(ap + 16);         // K {off+16..off+23}
      a[i] = u.v;
    }
#pragma unroll
    for (int j = 0; j < 4; ++j) {
      const __bf16* wp = W + (size_t)(n0 + j * 16 + l16) * Kdim + k0 + half * 8;
      V16Pack ub;
      ub.h[0] = *(const v8bf*)wp;               // K {off..off+7}
      ub.h[1] = *(const v8bf*)(wp + 16);        // K {off+16..off+23}
      v16bf b = ub.v;
#pragma unroll
      for (int i = 0; i < 4; ++i)
        acc[i][j] = wmma_bf16(a[i], b, acc[i][j]);
    }
    __syncthreads();                            // reads done before overwrite
  }

#pragma unroll
  for (int j = 0; j < 4; ++j) {
    const int col = n0 + j * 16 + l16;
    const float bv = bias ? bias[col] : 0.0f;
#pragma unroll
    for (int i = 0; i < 4; ++i) {
#pragma unroll
      for (int r = 0; r < 8; ++r) {
        const int row = m0 + i * 16 + r + half * 8;
        const float v = acc[i][j][r] + bv;
        if (OUT_BF16)
          ((__bf16*)Cout)[(size_t)row * Ndim + col] = (__bf16)v;
        else
          ((float*)Cout)[(size_t)row * Ndim + col] = v;
      }
    }
  }
}

// ---------------------------------------------------------------------------
// RoPE + layout shuffle. SCALE (1/sqrt(D)) is folded into Q here so the
// attention inner loop needs no score scaling.
//  q_lin/k_lin/v_lin: [B,S,E] bf16 (E = H*D)
//  Qo,Ko: [B,H,S,D] bf16 (rope applied; Qo pre-scaled)
//  Vt:    [B,H,D,S] bf16 (transposed so P*V B-operand loads are contiguous)
// ---------------------------------------------------------------------------
__global__ void rope_reshape_kernel(const __bf16* __restrict__ q_lin,
                                    const __bf16* __restrict__ k_lin,
                                    const __bf16* __restrict__ v_lin,
                                    const float* __restrict__ cosp,
                                    const float* __restrict__ sinp,
                                    __bf16* __restrict__ Qo,
                                    __bf16* __restrict__ Ko,
                                    __bf16* __restrict__ Vt) {
  const size_t i = (size_t)blockIdx.x * blockDim.x + threadIdx.x;
  if (i >= (size_t)Mx * Ex) return;
  const int d = (int)(i % Dx);
  const int h = (int)((i / Dx) % Hx);
  const size_t bs = i / Ex;                     // b*S + s
  const int b = (int)(bs / Sx);
  const int s = (int)(bs % Sx);

  const float c  = cosp[bs * Dx + d];
  const float sn = sinp[bs * Dx + d];
  const size_t j = i - d + ((d < 32) ? (d + 32) : (d - 32));
  const float sgn = (d < 32) ? -1.0f : 1.0f;

  const float qv = (float)q_lin[i], qr = (float)q_lin[j];
  const float kv = (float)k_lin[i], kr = (float)k_lin[j];
  const float vv = (float)v_lin[i];

  const size_t oidx = (((size_t)b * Hx + h) * Sx + s) * Dx + d;
  Qo[oidx] = (__bf16)((qv * c + sgn * qr * sn) * SCALE);
  Ko[oidx] = (__bf16)(kv * c + sgn * kr * sn);
  Vt[(((size_t)b * Hx + h) * Dx + d) * Sx + s] = (__bf16)vv;
}

// ---------------------------------------------------------------------------
// Flash attention, LDS-shared K/V with double-buffered async staging:
//  One workgroup (8 waves) owns 128 query rows of one (b,h). Per 32-key
//  step the K tile (32x64) and V^T tile (64x32) are async-staged into the
//  ping-pong LDS buffers; tile i+1's loads overlap tile i's WMMAs.
//  Per wave: 4 WMMAs Q*K^T, online softmax with cross-lane max only (the
//  row-sum is kept as per-lane partials -- rescale factor f is uniform per
//  16-lane row group -- and reduced once in the epilogue), P bounced through
//  per-wave LDS (C-layout -> A-layout), 4 WMMAs P*V.
// ---------------------------------------------------------------------------
__global__ __launch_bounds__(256)
void attn_kernel(const __bf16* __restrict__ Q,   // [B,H,S,D], pre-scaled
                 const __bf16* __restrict__ K,   // [B,H,S,D]
                 const __bf16* __restrict__ Vt,  // [B,H,D,S]
                 __bf16* __restrict__ O) {       // [B,S,E]
  __shared__ __bf16 Kt[2][32][64];               // 8KB ping-pong: keys x d
  __shared__ __bf16 Vs[2][64][32];               // 8KB ping-pong: d x keys
  __shared__ __bf16 Pb[8][16][32];               // 8KB per-wave P tiles

  const int tid  = threadIdx.x;
  const int lane = tid & 31;
  const int wid  = tid >> 5;
  const int nq   = Sx / 128;                     // 16 query blocks per (b,h)
  const int blk  = blockIdx.x;
  const int b    = blk / (Hx * nq);
  const int rem  = blk % (Hx * nq);
  const int h    = rem / nq;
  const int qb   = rem % nq;
  const int half = lane >> 4;
  const int l16  = lane & 15;
  const int q0   = qb * 128 + wid * 16;          // this wave's query rows

  const __bf16* qbase = Q  + (((size_t)b * Hx + h) * Sx + q0) * Dx;
  const __bf16* kbase = K  + (((size_t)b * Hx + h) * Sx) * Dx;
  const __bf16* vbase = Vt + (((size_t)b * Hx + h) * Dx) * Sx;

  // staging maps: one 16B chunk per thread per tile
  const int krow = tid >> 3, kch = (tid & 7) * 8;  // Kt: 32 x (64/8)
  const int vrow = tid >> 2, vch = (tid & 3) * 8;  // Vs: 64 x (32/8)
  const __bf16* kgp = kbase + (size_t)krow * Dx + kch;
  const __bf16* vgp = vbase + (size_t)vrow * Sx + vch;

  // Q tile as two 16x32 A-matrices (d 0..31 and 32..63); loop-invariant.
  v16bf a[2];
#pragma unroll
  for (int t = 0; t < 2; ++t) {
    const __bf16* ap = qbase + (size_t)l16 * Dx + t * 32 + half * 8;
    V16Pack u;
    u.h[0] = *(const v8bf*)ap;
    u.h[1] = *(const v8bf*)(ap + 16);
    a[t] = u.v;
  }

  float m[8], lp[8];
  v8f acc[4] = {};
#pragma unroll
  for (int r = 0; r < 8; ++r) { m[r] = -1e30f; lp[r] = 0.0f; }

  const int NIT = Sx / 32;
  stage16(kgp, &Kt[0][krow][kch]);               // prologue: tile 0
  stage16(vgp, &Vs[0][vrow][vch]);

  for (int it = 0; it < NIT; ++it) {
    const int kb  = it << 5;
    const int cur = it & 1;
    if (it + 1 < NIT) {                          // overlap next tile's loads
      stage16(kgp + (size_t)(kb + 32) * Dx, &Kt[cur ^ 1][krow][kch]);
      stage16(vgp + (kb + 32),              &Vs[cur ^ 1][vrow][vch]);
      async_wait<2>();                           // current tiles resident
    } else {
      async_wait<0>();
    }
    __syncthreads();

    // ---- scores: two 16x16 C tiles over 32 keys (Q pre-scaled) ----
    v8f sc[2] = {};
#pragma unroll
    for (int c = 0; c < 2; ++c) {
#pragma unroll
      for (int t = 0; t < 2; ++t) {
        const __bf16* kp = &Kt[cur][c * 16 + l16][t * 32 + half * 8];
        V16Pack ub;
        ub.h[0] = *(const v8bf*)kp;
        ub.h[1] = *(const v8bf*)(kp + 16);
        v16bf bk = ub.v;
        sc[c] = wmma_bf16(a[t], bk, sc[c]);
      }
    }

    // ---- online softmax: cross-lane max only; sum kept per-lane ----
    float f[8];
#pragma unroll
    for (int r = 0; r < 8; ++r) {
      const float s0 = sc[0][r];
      const float s1 = sc[1][r];
      float t = fmaxf(s0, s1);
      t = fmaxf(t, __shfl_xor(t, 1));
      t = fmaxf(t, __shfl_xor(t, 2));
      t = fmaxf(t, __shfl_xor(t, 4));
      t = fmaxf(t, __shfl_xor(t, 8));
      const float mn = fmaxf(m[r], t);
      f[r] = __expf(m[r] - mn);
      m[r] = mn;
      const float p0 = __expf(s0 - mn);
      const float p1 = __expf(s1 - mn);
      sc[0][r] = p0;
      sc[1][r] = p1;
      lp[r] = lp[r] * f[r] + (p0 + p1);          // per-lane partial row sum
    }
#pragma unroll
    for (int c2 = 0; c2 < 4; ++c2)
#pragma unroll
      for (int r = 0; r < 8; ++r) acc[c2][r] *= f[r];

    // ---- P (C-layout) -> per-wave LDS -> A-layout ----
#pragma unroll
    for (int r = 0; r < 8; ++r) {
      const int row = r + half * 8;
      Pb[wid][row][l16]      = (__bf16)sc[0][r];
      Pb[wid][row][16 + l16] = (__bf16)sc[1][r];
    }
    asm volatile("" ::: "memory");   // keep DS store->load order at IR level
    v16bf pa;
    {
      const __bf16* lpt = &Pb[wid][l16][half * 8];
      V16Pack u;
      u.h[0] = *(const v8bf*)lpt;
      u.h[1] = *(const v8bf*)(lpt + 16);
      pa = u.v;
    }

    // ---- O += P * V from the shared V^T tile ----
#pragma unroll
    for (int c2 = 0; c2 < 4; ++c2) {
      const __bf16* vp = &Vs[cur][c2 * 16 + l16][half * 8];
      V16Pack ub;
      ub.h[0] = *(const v8bf*)vp;
      ub.h[1] = *(const v8bf*)(vp + 16);
      v16bf bv = ub.v;
      acc[c2] = wmma_bf16(pa, bv, acc[c2]);
    }

    __syncthreads();                 // all waves done before next staging
  }

  // ---- one deferred cross-lane sum per row, then normalize + emit ----
  float linv[8];
#pragma unroll
  for (int r = 0; r < 8; ++r) {
    float rs = lp[r];
    rs += __shfl_xor(rs, 1);
    rs += __shfl_xor(rs, 2);
    rs += __shfl_xor(rs, 4);
    rs += __shfl_xor(rs, 8);
    linv[r] = 1.0f / rs;
  }
#pragma unroll
  for (int c2 = 0; c2 < 4; ++c2) {
    const int col = h * 64 + c2 * 16 + l16;
#pragma unroll
    for (int r = 0; r < 8; ++r) {
      const int srow = q0 + r + half * 8;
      O[((size_t)b * Sx + srow) * Ex + col] = (__bf16)(acc[c2][r] * linv[r]);
    }
  }
}

// ---------------------------------------------------------------------------
// Host-side orchestration
// ---------------------------------------------------------------------------
extern "C" void kernel_launch(void* const* d_in, const int* in_sizes, int n_in,
                              void* d_out, int out_size, void* d_ws, size_t ws_size,
                              hipStream_t stream) {
  const float* hidden = (const float*)d_in[0];
  const float* cosp   = (const float*)d_in[1];
  const float* sinp   = (const float*)d_in[2];
  const float* Wq     = (const float*)d_in[3];
  const float* bq     = (const float*)d_in[4];
  const float* Wk     = (const float*)d_in[5];
  const float* bk     = (const float*)d_in[6];
  const float* Wv     = (const float*)d_in[7];
  const float* bv     = (const float*)d_in[8];
  const float* Wo     = (const float*)d_in[9];
  const float* bo     = (const float*)d_in[10];
  float* out = (float*)d_out;

  char* ws = (char*)d_ws;
  size_t off = 0;
  auto carve = [&](size_t bytes) -> char* {
    char* p = ws + off;
    off += (bytes + 255) & ~(size_t)255;
    return p;
  };

  const size_t actBytes = (size_t)Mx * Ex * sizeof(__bf16);   // 16 MB
  const size_t wBytes   = (size_t)Ex * Ex * sizeof(__bf16);   //  2 MB

  __bf16* h_bf  = (__bf16*)carve(actBytes);  // reused later as attn output
  __bf16* wq_bf = (__bf16*)carve(wBytes);
  __bf16* wk_bf = (__bf16*)carve(wBytes);
  __bf16* wv_bf = (__bf16*)carve(wBytes);
  __bf16* wo_bf = (__bf16*)carve(wBytes);
  __bf16* q_lin = (__bf16*)carve(actBytes);
  __bf16* k_lin = (__bf16*)carve(actBytes);
  __bf16* v_lin = (__bf16*)carve(actBytes);
  __bf16* q_bf  = (__bf16*)carve(actBytes);  // [B,H,S,D]
  __bf16* k_bf  = (__bf16*)carve(actBytes);  // [B,H,S,D]
  __bf16* v_t   = (__bf16*)carve(actBytes);  // [B,H,D,S]
  __bf16* attn_bf = h_bf;                    // alias: hidden_bf dead by then

  // 1) fp32 -> bf16
  {
    const int nAct = Mx * Ex, nW = Ex * Ex;
    f32_to_bf16_kernel<<<(nAct + 255) / 256, 256, 0, stream>>>(hidden, h_bf, nAct);
    f32_to_bf16_kernel<<<(nW + 255) / 256, 256, 0, stream>>>(Wq, wq_bf, nW);
    f32_to_bf16_kernel<<<(nW + 255) / 256, 256, 0, stream>>>(Wk, wk_bf, nW);
    f32_to_bf16_kernel<<<(nW + 255) / 256, 256, 0, stream>>>(Wv, wv_bf, nW);
    f32_to_bf16_kernel<<<(nW + 255) / 256, 256, 0, stream>>>(Wo, wo_bf, nW);
  }

  // 2) QKV projections: waves = (M/64)*(E/64) = 2048 -> 256 blocks of 8 waves
  const int gemmBlocks = (Mx / 64) * (Ex / 64) / 8;
  gemm_bf16_nt_kernel<true><<<gemmBlocks, 256, 0, stream>>>(
      h_bf, wq_bf, bq, q_lin, Mx, Ex, Ex);
  gemm_bf16_nt_kernel<true><<<gemmBlocks, 256, 0, stream>>>(
      h_bf, wk_bf, bk, k_lin, Mx, Ex, Ex);
  gemm_bf16_nt_kernel<true><<<gemmBlocks, 256, 0, stream>>>(
      h_bf, wv_bf, bv, v_lin, Mx, Ex, Ex);

  // 3) RoPE + layout ([B,H,S,D] for Q/K, [B,H,D,S] for V); Q pre-scaled
  {
    const size_t n = (size_t)Mx * Ex;
    rope_reshape_kernel<<<(unsigned)((n + 255) / 256), 256, 0, stream>>>(
        q_lin, k_lin, v_lin, cosp, sinp, q_bf, k_bf, v_t);
  }

  // 4) attention: one block per (b, h, 128-query block) = 1024 blocks
  attn_kernel<<<Bx * Hx * (Sx / 128), 256, 0, stream>>>(q_bf, k_bf, v_t, attn_bf);

  // 5) output projection -> fp32 d_out
  gemm_bf16_nt_kernel<false><<<gemmBlocks, 256, 0, stream>>>(
      attn_bf, wo_bf, bo, out, Mx, Ex, Ex);
}
